// VoxelGridImplicitFunction_8392366096536
// MI455X (gfx1250) — matrix operand order repaired and checked by reading, not compiled, not run on hardware
//
#include <hip/hip_runtime.h>

// ---------------- problem constants ----------------
#define BB   2
#define RR   2048
#define PP   64
#define NPTS (BB * RR * PP)   // 262144
#define GRES 128
#define GC   32
#define HID  256

#define KD   416   // density feature dim (13 * 32)
#define KC_  448   // color feature dim padded (443 -> 448 = 14 * 32)
#define KCD  13    // K chunks density layer1
#define KCC  14    // K chunks color layer1
#define KCH  8     // K chunks for 256-wide layers

typedef __attribute__((ext_vector_type(16))) _Float16 v16h;
typedef __attribute__((ext_vector_type(8)))  _Float16 v8h;
typedef __attribute__((ext_vector_type(2)))  _Float16 v2h;
typedef __attribute__((ext_vector_type(8)))  float    v8f;

// ---------------- LDS layout (per wave) ----------------
#define WAVES        4
#define FEAT_HALVES  (KC_ * 16)          // 7168 halves = 14336 B (A tile, also reused as h2)
#define H_HALVES     (HID * 16)          // 4096 halves = 8192 B  (h1 tile)
#define WB_FEAT      (FEAT_HALVES * 2)
#define WB_H         (H_HALVES * 2)
#define WB_OCC       64
#define WB_WAVE      (WB_FEAT + WB_H + WB_OCC)   // 22592 (16B aligned)
#define SMEM_BYTES   (WAVES * WB_WAVE)           // 90368

// ---------------- workspace layout (bytes) ----------------
#define WS_WD1F 0
#define WS_WD2F (WS_WD1F + (size_t)KCD * 16 * 32 * 16 * 2)  // 212992
#define WS_WD3F (WS_WD2F + (size_t)KCH * 16 * 32 * 16 * 2)  // +131072
#define WS_WC1F (WS_WD3F + (size_t)KCH * 1  * 32 * 16 * 2)  // +8192
#define WS_WC2F (WS_WC1F + (size_t)KCC * 16 * 32 * 16 * 2)  // +229376
#define WS_WC3F (WS_WC2F + (size_t)KCH * 16 * 32 * 16 * 2)  // +131072

// =====================================================================
// Weight packing: W[K x N] (fp32, row-major) -> f16 WMMA B fragments.
// Fragment f = c*NT + t (c = K chunk of 32, t = 16-col tile).
// Per-lane 16 contiguous halves: lane L holds column n = t*16 + (L&15),
// K rows kb..kb+7, kb+16..kb+23 with kb = c*32 + (L>=16 ? 8 : 0)  (ISA B layout).
// =====================================================================
__global__ void pack_weights_kernel(const float* __restrict__ W, int K, int N,
                                    _Float16* __restrict__ out, int KC, int NT)
{
    int i = blockIdx.x * blockDim.x + threadIdx.x;
    int total = KC * NT * 32;
    if (i >= total) return;
    int L = i & 31;
    int f = i >> 5;
    int t = f % NT;
    int c = f / NT;
    int n  = t * 16 + (L & 15);
    int kb = c * 32 + ((L >= 16) ? 8 : 0);
    _Float16* o = out + (size_t)i * 16;
#pragma unroll
    for (int h = 0; h < 16; ++h) {
        int k = kb + (h < 8 ? h : h + 8);
        float v = (k < K && n < N) ? W[(size_t)k * N + n] : 0.f;
        o[h] = (_Float16)v;
    }
}

// =====================================================================
// GEMM layers (wave-scope): A tile in LDS (row-major f16), B fragments
// pre-packed in global, C-init = bias, f16 WMMA 16x16x32 accumulation.
// =====================================================================
template<int KC>
__device__ __forceinline__ void mlp_layer_relu(
    const _Float16* __restrict__ Ab, int astride,
    const _Float16* __restrict__ wfrag, const float* __restrict__ bias,
    _Float16* __restrict__ Hout, int lane)
{
    const int off8 = (lane >= 16) ? 8 : 0;
    const _Float16* arow = Ab + (lane & 15) * astride + off8;  // A: M = lane&15
#pragma unroll 1
    for (int t = 0; t < 16; ++t) {
        float bn = bias[t * 16 + (lane & 15)];                 // C/D: N = lane&15
        v8f acc = {bn, bn, bn, bn, bn, bn, bn, bn};
#pragma unroll
        for (int c = 0; c < KC; ++c) {
            v16h a, b;
            ((v8h*)&a)[0] = *(const v8h*)(arow + c * 32);
            ((v8h*)&a)[1] = *(const v8h*)(arow + c * 32 + 16);
            const _Float16* bp = wfrag + (((size_t)(c * 16 + t)) * 32 + lane) * 16;
            ((v8h*)&b)[0] = *(const v8h*)(bp);
            ((v8h*)&b)[1] = *(const v8h*)(bp + 8);
            acc = __builtin_amdgcn_wmma_f32_16x16x32_f16(
                false, a, false, b, (short)0, acc, false, false);
        }
#pragma unroll
        for (int j = 0; j < 8; ++j) {                          // D: row m = j + off8
            float v = acc[j] > 0.f ? acc[j] : 0.f;
            Hout[(j + off8) * HID + t * 16 + (lane & 15)] = (_Float16)v;
        }
    }
}

template<int DOUT>
__device__ __forceinline__ void mlp_layer_out(
    const _Float16* __restrict__ Ab, int astride,
    const _Float16* __restrict__ wfrag, const float* __restrict__ bias,
    const float* __restrict__ occm, float* __restrict__ outp,
    int pbase, int lane)
{
    const int col  = lane & 15;
    const int off8 = (lane >= 16) ? 8 : 0;
    const _Float16* arow = Ab + (lane & 15) * astride + off8;
    float bn = (col < DOUT) ? bias[col] : 0.f;
    v8f acc = {bn, bn, bn, bn, bn, bn, bn, bn};
#pragma unroll
    for (int c = 0; c < KCH; ++c) {
        v16h a, b;
        ((v8h*)&a)[0] = *(const v8h*)(arow + c * 32);
        ((v8h*)&a)[1] = *(const v8h*)(arow + c * 32 + 16);
        const _Float16* bp = wfrag + ((size_t)c * 32 + lane) * 16;
        ((v8h*)&b)[0] = *(const v8h*)(bp);
        ((v8h*)&b)[1] = *(const v8h*)(bp + 8);
        acc = __builtin_amdgcn_wmma_f32_16x16x32_f16(
            false, a, false, b, (short)0, acc, false, false);
    }
    if (col < DOUT) {
#pragma unroll
        for (int j = 0; j < 8; ++j) {
            int mm = j + off8;
            outp[(size_t)(pbase + mm) * DOUT + col] = acc[j] * occm[mm];
        }
    }
}

// =====================================================================
// Embed 16 channels of one grid for one point; write harmonic features
// (sin | cos | identity, NH=6) as f16 into the LDS A-tile row.
// sin/cos at freqs 2^f computed with one sin/cos + double-angle chain;
// pairs packed into 32-bit LDS stores (offsets ch*12 B are 4-aligned).
// =====================================================================
__device__ __forceinline__ void embed_grid16(
    const float* __restrict__ grid, int ch0,
    int x0, int y0, int z0, float fx, float fy, float fz,
    _Float16* __restrict__ featRow)
{
    float v[16];
#pragma unroll
    for (int i = 0; i < 16; ++i) v[i] = 0.f;
#pragma unroll
    for (int cn = 0; cn < 8; ++cn) {
        const int cx = x0 + (cn & 1);
        const int cy = y0 + ((cn >> 1) & 1);
        const int cz = z0 + (cn >> 2);
        const float wx = (cn & 1)        ? fx : 1.f - fx;
        const float wy = ((cn >> 1) & 1) ? fy : 1.f - fy;
        const float wz = (cn >> 2)       ? fz : 1.f - fz;
        const float w = wx * wy * wz;
        const float* gp = grid + (((size_t)cz * GRES + cy) * GRES + cx) * GC + ch0;
        const float4 q0 = ((const float4*)gp)[0];
        const float4 q1 = ((const float4*)gp)[1];
        const float4 q2 = ((const float4*)gp)[2];
        const float4 q3 = ((const float4*)gp)[3];
        v[0]  += w * q0.x; v[1]  += w * q0.y; v[2]  += w * q0.z; v[3]  += w * q0.w;
        v[4]  += w * q1.x; v[5]  += w * q1.y; v[6]  += w * q1.z; v[7]  += w * q1.w;
        v[8]  += w * q2.x; v[9]  += w * q2.y; v[10] += w * q2.z; v[11] += w * q2.w;
        v[12] += w * q3.x; v[13] += w * q3.y; v[14] += w * q3.z; v[15] += w * q3.w;
    }
#pragma unroll 1
    for (int i = 0; i < 16; ++i) {
        const int ch = ch0 + i;
        const float x = v[i];
        float s = __sinf(x), c = __cosf(x);
        float sv[6], cv[6];
#pragma unroll
        for (int f = 0; f < 6; ++f) {
            sv[f] = s; cv[f] = c;
            const float s2 = 2.f * s * c;        // sin(2t)
            const float c2 = 1.f - 2.f * s * s;  // cos(2t)
            s = s2; c = c2;
        }
        v2h* sp = (v2h*)(featRow + ch * 6);
        v2h* cp = (v2h*)(featRow + 192 + ch * 6);
#pragma unroll
        for (int f = 0; f < 3; ++f) {
            v2h sa = {(_Float16)sv[2 * f], (_Float16)sv[2 * f + 1]};
            v2h ca = {(_Float16)cv[2 * f], (_Float16)cv[2 * f + 1]};
            sp[f] = sa;
            cp[f] = ca;
        }
        featRow[384 + ch] = (_Float16)x;
    }
}

// =====================================================================
// Fused kernel: trilerp + harmonic -> LDS A tiles -> WMMA MLPs -> out.
// 128 threads = 4 waves; each wave owns 16 points (one WMMA M tile).
// =====================================================================
__global__ __launch_bounds__(WAVES * 32) void nerf_fused_kernel(
    const float* __restrict__ origins,   const float* __restrict__ directions,
    const float* __restrict__ lengths,
    const float* __restrict__ gden,      const float* __restrict__ gcol,
    const float* __restrict__ gsca,
    const float* __restrict__ bd1, const float* __restrict__ bd2, const float* __restrict__ bd3,
    const float* __restrict__ bc1, const float* __restrict__ bc2, const float* __restrict__ bc3,
    const _Float16* __restrict__ wd1f, const _Float16* __restrict__ wd2f, const _Float16* __restrict__ wd3f,
    const _Float16* __restrict__ wc1f, const _Float16* __restrict__ wc2f, const _Float16* __restrict__ wc3f,
    float* __restrict__ out)
{
    extern __shared__ char smem_raw[];
    const int lane = threadIdx.x & 31;
    const int wave = threadIdx.x >> 5;
    _Float16* featA = (_Float16*)(smem_raw + wave * WB_WAVE);
    _Float16* hbuf  = featA + FEAT_HALVES;
    float*    occm  = (float*)(smem_raw + wave * WB_WAVE + WB_FEAT + WB_H);

    const int pbase = (blockIdx.x * WAVES + wave) * 16;
    const int m   = lane & 15;
    const int ch0 = (lane >> 4) * 16;    // channel half owned by this lane
    const int p   = pbase + m;
    const int ray = p / PP;              // b*RR + r
    const int s   = p - ray * PP;

    const float ox = origins[ray * 3 + 0], oy = origins[ray * 3 + 1], oz = origins[ray * 3 + 2];
    const float dx = directions[ray * 3 + 0], dy = directions[ray * 3 + 1], dz = directions[ray * 3 + 2];
    const float tl = lengths[ray * PP + s];
    const float px = ox + dx * tl, py = oy + dy * tl, pz = oz + dz * tl;

    const float xs = (px + 1.f) * 0.5f * (GRES - 1);
    const float ys = (py + 1.f) * 0.5f * (GRES - 1);
    const float zs = (pz + 1.f) * 0.5f * (GRES - 1);
    const int x0 = min(max((int)floorf(xs), 0), GRES - 2);
    const int y0 = min(max((int)floorf(ys), 0), GRES - 2);
    const int z0 = min(max((int)floorf(zs), 0), GRES - 2);
    const float fx = fminf(fmaxf(xs - (float)x0, 0.f), 1.f);
    const float fy = fminf(fmaxf(ys - (float)y0, 0.f), 1.f);
    const float fz = fminf(fmaxf(zs - (float)z0, 0.f), 1.f);

    // Prefetch color-grid corner lines now; consumed only after the whole
    // density MLP (hides HBM gather latency behind ~350 WMMAs).
#pragma unroll
    for (int cn = 0; cn < 8; ++cn) {
        const int cx = x0 + (cn & 1), cy = y0 + ((cn >> 1) & 1), cz = z0 + (cn >> 2);
        __builtin_prefetch(gcol + (((size_t)cz * GRES + cy) * GRES + cx) * GC + ch0, 0, 3);
    }

    _Float16* featRow = featA + m * KC_;

    // ---------------- density stream ----------------
    embed_grid16(gden, ch0, x0, y0, z0, fx, fy, fz, featRow);
    if (lane < 16) {  // scaffold occupancy (1 channel)
        float sv = 0.f;
#pragma unroll
        for (int cn = 0; cn < 8; ++cn) {
            const int cx = x0 + (cn & 1), cy = y0 + ((cn >> 1) & 1), cz = z0 + (cn >> 2);
            const float wx = (cn & 1) ? fx : 1.f - fx;
            const float wy = ((cn >> 1) & 1) ? fy : 1.f - fy;
            const float wz = (cn >> 2) ? fz : 1.f - fz;
            sv += wx * wy * wz * gsca[((size_t)cz * GRES + cy) * GRES + cx];
        }
        occm[m] = (sv > 0.f) ? 1.f : 0.f;
    }
    __syncthreads();
    mlp_layer_relu<KCD>(featA, KC_, wd1f, bd1, hbuf, lane);   // 416 -> 256
    __syncthreads();
    mlp_layer_relu<KCH>(hbuf, HID, wd2f, bd2, featA, lane);   // 256 -> 256 (reuse featA as h2)
    __syncthreads();
    mlp_layer_out<1>(featA, HID, wd3f, bd3, occm, out, pbase, lane);  // 256 -> 1
    __syncthreads();

    // ---------------- color stream ----------------
    embed_grid16(gcol, ch0, x0, y0, z0, fx, fy, fz, featRow);
    if (lane < 16) {  // normalized view-dir harmonic (27 values) + pad to 448
        const float inv = 1.f / sqrtf(dx * dx + dy * dy + dz * dz);
        const float nd[3] = {dx * inv, dy * inv, dz * inv};
#pragma unroll
        for (int comp = 0; comp < 3; ++comp) {
            float s = __sinf(nd[comp]), c = __cosf(nd[comp]);
            float sv[4], cv[4];
#pragma unroll
            for (int f = 0; f < 4; ++f) {
                sv[f] = s; cv[f] = c;
                const float s2 = 2.f * s * c;
                const float c2 = 1.f - 2.f * s * s;
                s = s2; c = c2;
            }
            v2h* sp = (v2h*)(featRow + KD + comp * 4);        // 8B-aligned
            v2h* cp = (v2h*)(featRow + KD + 12 + comp * 4);   // 4B-aligned
#pragma unroll
            for (int f = 0; f < 2; ++f) {
                v2h sa = {(_Float16)sv[2 * f], (_Float16)sv[2 * f + 1]};
                v2h ca = {(_Float16)cv[2 * f], (_Float16)cv[2 * f + 1]};
                sp[f] = sa;
                cp[f] = ca;
            }
            featRow[KD + 24 + comp] = (_Float16)nd[comp];
        }
#pragma unroll
        for (int k = 443; k < KC_; ++k) featRow[k] = (_Float16)0.f;
    }
    __syncthreads();
    mlp_layer_relu<KCC>(featA, KC_, wc1f, bc1, hbuf, lane);   // 443(448) -> 256
    __syncthreads();
    mlp_layer_relu<KCH>(hbuf, HID, wc2f, bc2, featA, lane);   // 256 -> 256
    __syncthreads();
    mlp_layer_out<3>(featA, HID, wc3f, bc3, occm, out + NPTS, pbase, lane);  // 256 -> 3
}

// =====================================================================
extern "C" void kernel_launch(void* const* d_in, const int* in_sizes, int n_in,
                              void* d_out, int out_size, void* d_ws, size_t ws_size,
                              hipStream_t stream) {
    const float* origins    = (const float*)d_in[0];
    const float* directions = (const float*)d_in[1];
    const float* lengths    = (const float*)d_in[2];
    const float* gden       = (const float*)d_in[3];
    const float* gcol       = (const float*)d_in[4];
    const float* gsca       = (const float*)d_in[5];
    const float* wd1 = (const float*)d_in[6];  const float* bd1 = (const float*)d_in[7];
    const float* wd2 = (const float*)d_in[8];  const float* bd2 = (const float*)d_in[9];
    const float* wd3 = (const float*)d_in[10]; const float* bd3 = (const float*)d_in[11];
    const float* wc1 = (const float*)d_in[12]; const float* bc1 = (const float*)d_in[13];
    const float* wc2 = (const float*)d_in[14]; const float* bc2 = (const float*)d_in[15];
    const float* wc3 = (const float*)d_in[16]; const float* bc3 = (const float*)d_in[17];

    char* ws = (char*)d_ws;
    _Float16* wd1f = (_Float16*)(ws + WS_WD1F);
    _Float16* wd2f = (_Float16*)(ws + WS_WD2F);
    _Float16* wd3f = (_Float16*)(ws + WS_WD3F);
    _Float16* wc1f = (_Float16*)(ws + WS_WC1F);
    _Float16* wc2f = (_Float16*)(ws + WS_WC2F);
    _Float16* wc3f = (_Float16*)(ws + WS_WC3F);

    auto frags = [](int KC, int NT) { return KC * NT * 32; };
    auto blocks = [](int n) { return (n + 255) / 256; };

    pack_weights_kernel<<<blocks(frags(KCD, 16)), 256, 0, stream>>>(wd1, 416, HID, wd1f, KCD, 16);
    pack_weights_kernel<<<blocks(frags(KCH, 16)), 256, 0, stream>>>(wd2, HID, HID, wd2f, KCH, 16);
    pack_weights_kernel<<<blocks(frags(KCH, 1)),  256, 0, stream>>>(wd3, HID, 1,   wd3f, KCH, 1);
    pack_weights_kernel<<<blocks(frags(KCC, 16)), 256, 0, stream>>>(wc1, 443, HID, wc1f, KCC, 16);
    pack_weights_kernel<<<blocks(frags(KCH, 16)), 256, 0, stream>>>(wc2, HID, HID, wc2f, KCH, 16);
    pack_weights_kernel<<<blocks(frags(KCH, 1)),  256, 0, stream>>>(wc3, HID, 3,   wc3f, KCH, 1);

    const int nblocks = NPTS / (16 * WAVES);  // 4096
    nerf_fused_kernel<<<nblocks, WAVES * 32, SMEM_BYTES, stream>>>(
        origins, directions, lengths, gden, gcol, gsca,
        bd1, bd2, bd3, bc1, bc2, bc3,
        wd1f, wd2f, wd3f, wc1f, wc2f, wc3f,
        (float*)d_out);
}
